// TextLSTM_63522566308160
// MI455X (gfx1250) — compile-verified, matrix operand order checked by reading, not run on hardware
//
#include <hip/hip_runtime.h>
#include <hip/hip_bf16.h>

// Problem dimensions (match reference).
#define N_CLASS 32000
#define EMB     1024
#define HID     1024
#define BATCH   256
#define SEQ     256
#define KCAT    (EMB + HID)   // 2048
#define NGATE   (4 * HID)     // 4096

#define KC      64            // K-chunk depth staged per barrier
#define ROWP    72            // padded LDS row stride (halves): r*36 dwords % 64 banks all-distinct
#define PROJ_P  1032          // padded LDS row stride (halves) for 1024-K A panel

// CDNA5 async global->LDS staging (ASYNCcnt) when the toolchain exposes it.
#if __has_builtin(__builtin_amdgcn_global_load_async_to_lds_b128) && \
    __has_builtin(__builtin_amdgcn_s_wait_asynccnt)
#define ASYNC_STAGE 1
#endif

#define AS_GLOBAL __attribute__((address_space(1)))
#define AS_LDS    __attribute__((address_space(3)))

typedef __attribute__((ext_vector_type(16))) _Float16 v16h;
typedef __attribute__((ext_vector_type(8)))  _Float16 v8h;
typedef __attribute__((ext_vector_type(4)))  _Float16 v4h;
typedef __attribute__((ext_vector_type(8)))  float    v8f;
typedef __attribute__((ext_vector_type(4)))  int      v4i;

// ---------------------------------------------------------------------------
// WMMA 16x16x32 f16 fragment load from a row-major array (16B-aligned rows).
// 16-bit A layout (ISA 7.12.2): lane -> {kb=(lane/16)*8, r=lane%16};
// VGPR0..3 hold K=kb+0..7, VGPR4..7 hold K=kb+16..23.
// Works for global or LDS pointers (inlined -> addrspace inferred).
// ---------------------------------------------------------------------------
__device__ __forceinline__ v16h load_frag16(const _Float16* __restrict__ base,
                                            int stride) {
  const int lane = threadIdx.x & 31;
  const int r    = lane & 15;
  const int kb   = (lane >> 4) << 3;    // 0 or 8
  const _Float16* p = base + (size_t)r * stride + kb;
  v8h lo = *reinterpret_cast<const v8h*>(p);        // K = kb + 0..7
  v8h hi = *reinterpret_cast<const v8h*>(p + 16);   // K = kb + 16..23
  v16h out;
#pragma unroll
  for (int i = 0; i < 8; ++i) { out[i] = lo[i]; out[i + 8] = hi[i]; }
  return out;
}

__device__ __forceinline__ float sigmoidf_(float x) {
  return 1.0f / (1.0f + __expf(-x));
}

__device__ __forceinline__ void stage_fence() {
#ifdef ASYNC_STAGE
  __builtin_amdgcn_s_wait_asynccnt(0);
#endif
  __syncthreads();
}

// Copy 64B global -> LDS. Async path: 4x GLOBAL_LOAD_ASYNC_TO_LDS_B128
// (ASYNCcnt, no VGPR round-trip). Fallback: VGPR-mediated b128 copies.
__device__ __forceinline__ void stage_copy64(const _Float16* __restrict__ src,
                                             _Float16* __restrict__ dst) {
#ifdef ASYNC_STAGE
  AS_GLOBAL v4i* gp = (AS_GLOBAL v4i*)src;
  AS_LDS    v4i* lp = (AS_LDS v4i*)dst;
  __builtin_amdgcn_global_load_async_to_lds_b128(gp,     lp,     0, 0);
  __builtin_amdgcn_global_load_async_to_lds_b128(gp + 1, lp + 1, 0, 0);
  __builtin_amdgcn_global_load_async_to_lds_b128(gp + 2, lp + 2, 0, 0);
  __builtin_amdgcn_global_load_async_to_lds_b128(gp + 3, lp + 3, 0, 0);
#else
#pragma unroll
  for (int q = 0; q < 4; ++q) {
    v8h v = *reinterpret_cast<const v8h*>(src + q * 8);
    *reinterpret_cast<v8h*>(dst + q * 8) = v;
  }
#endif
}

// ---------------------------------------------------------------------------
// Packing / utility kernels
// ---------------------------------------------------------------------------

// E[t][b][:] = (f16) C[X[b][t]][:]   (time-major; one plane per step)
__global__ void embed_kernel(const int* __restrict__ X,
                             const float* __restrict__ C,
                             _Float16* __restrict__ E) {
  const int t = blockIdx.x;
  const int b = blockIdx.y;
  const int tok = X[(size_t)b * SEQ + t];
  const float4* src = (const float4*)(C + (size_t)tok * EMB);
  _Float16* dst = E + ((size_t)t * BATCH + b) * EMB;
  const int e = threadIdx.x;        // 256 threads, 4 elems each
  float4 v = src[e];
  v4h o = { (_Float16)v.x, (_Float16)v.y, (_Float16)v.z, (_Float16)v.w };
  *reinterpret_cast<v4h*>(dst + e * 4) = o;
}

// Wcat[n][k], n = gate*HID + j, source W_gate[j][k]  (gate order i,f,o,g)
__global__ void pack_gates_kernel(const float* __restrict__ Wi,
                                  const float* __restrict__ Wf,
                                  const float* __restrict__ Wo,
                                  const float* __restrict__ Wg,
                                  _Float16* __restrict__ Wcat) {
  size_t idx = (size_t)blockIdx.x * blockDim.x + threadIdx.x;
  if (idx >= (size_t)NGATE * KCAT) return;
  const int n    = (int)(idx / KCAT);
  const int k    = (int)(idx % KCAT);
  const int gate = n >> 10;
  const int j    = n & (HID - 1);
  const float* src = (gate == 0) ? Wi : (gate == 1) ? Wf : (gate == 2) ? Wo : Wg;
  Wcat[idx] = (_Float16)src[(size_t)j * KCAT + k];
}

__global__ void cvt_f32_to_f16_kernel(const float* __restrict__ src,
                                      _Float16* __restrict__ dst, size_t n) {
  size_t i = (size_t)blockIdx.x * blockDim.x + threadIdx.x;
  if (i < n) dst[i] = (_Float16)src[i];
}

__global__ void zero_f16_kernel(_Float16* __restrict__ p, size_t n) {
  size_t i = (size_t)blockIdx.x * blockDim.x + threadIdx.x;
  if (i < n) p[i] = (_Float16)0.0f;
}

__global__ void zero_f32_kernel(float* __restrict__ p, size_t n) {
  size_t i = (size_t)blockIdx.x * blockDim.x + threadIdx.x;
  if (i < n) p[i] = 0.0f;
}

// hsum += (float)h
__global__ void accum_h_kernel(float* __restrict__ hsum,
                               const _Float16* __restrict__ h, size_t n) {
  size_t i = (size_t)blockIdx.x * blockDim.x + threadIdx.x;
  if (i < n) hsum[i] += (float)h[i];
}

// ---------------------------------------------------------------------------
// One LSTM time step, fused cell update.
//   gates = [E_t | h_{t-1}] @ Wcat^T   (M=256, K=2048, N=4096 in 4 gate panels)
// Block: 128 threads = 4 waves; each wave owns 32(b) x 16(j) x 4 gates.
// The 4-gate 64-K weight chunk (8KB) is shared by all waves: staged into LDS
// with double buffering via GLOBAL_LOAD_ASYNC_TO_LDS_B128 (ASYNCcnt), fenced
// with s_wait_asynccnt + barrier. K-loop split at EMB so each phase has a
// constant A-plane base pointer (no per-iteration selects).
// Grid: (BATCH/128, HID/16). Hidden state double-buffered across steps.
// ---------------------------------------------------------------------------
__global__ void __launch_bounds__(128)
lstm_step_kernel(const _Float16* __restrict__ Ecur,  // [BATCH][EMB]
                 const _Float16* __restrict__ Hin,   // [BATCH][HID]
                 _Float16* __restrict__ Hout,        // [BATCH][HID]
                 float* __restrict__ Cst,            // [BATCH][HID]
                 const _Float16* __restrict__ Wcat,  // [NGATE][KCAT]
                 const float* __restrict__ bi, const float* __restrict__ bf,
                 const float* __restrict__ bo, const float* __restrict__ bg) {
  __shared__ _Float16 lB[2][64 * ROWP];   // [buf][(gate*16+j) * ROWP + k], 2 x 9216 B

  const int wave = threadIdx.x >> 5;
  const int lane = threadIdx.x & 31;
  const int b0 = blockIdx.x * 128 + wave * 32;   // this wave: b0 .. b0+31
  const int j0 = blockIdx.y * 16;

  // Stage one 64-K weight chunk (64 rows x 64 halves) into lB[buf].
  // 128 threads x 64B each: thread -> (row = tid/2, seg = (tid&1)*32 halves).
  auto stage = [&](int buf, int k0) {
    const int row  = threadIdx.x >> 1;          // 0..63 = gate*16 + j
    const int seg  = (threadIdx.x & 1) << 5;    // 0 or 32 (halves)
    const int gate = row >> 4;
    const int j    = row & 15;
    const _Float16* src =
        Wcat + (size_t)(gate * HID + j0 + j) * KCAT + k0 + seg;
    _Float16* dst = &lB[buf][row * ROWP + seg];
    stage_copy64(src, dst);
  };

  v8f acc[4][2];
#pragma unroll
  for (int g = 0; g < 4; ++g)
#pragma unroll
    for (int mt = 0; mt < 2; ++mt) acc[g][mt] = (v8f){};

  // One 64-K chunk: 2 sub-chunks x (2 A frags + 4 B frags) -> 16 WMMAs.
  auto compute = [&](int buf, const _Float16* __restrict__ Aplane, int ka) {
#pragma unroll
    for (int sub = 0; sub < 2; ++sub) {
      const int kk = ka + sub * 32;
      v16h a0 = load_frag16(Aplane + (size_t)(b0 +  0) * 1024 + kk, 1024);
      v16h a1 = load_frag16(Aplane + (size_t)(b0 + 16) * 1024 + kk, 1024);
#pragma unroll
      for (int g = 0; g < 4; ++g) {
        v16h bm = load_frag16(&lB[buf][g * 16 * ROWP + sub * 32], ROWP);
        acc[g][0] = __builtin_amdgcn_wmma_f32_16x16x32_f16(
            false, a0, false, bm, (short)0, acc[g][0], false, false);
        acc[g][1] = __builtin_amdgcn_wmma_f32_16x16x32_f16(
            false, a1, false, bm, (short)0, acc[g][1], false, false);
      }
    }
  };

  stage(0, 0);
  stage_fence();

  // Phase 1: K in [0, EMB) -> A from the embedding plane.
  for (int k0 = 0; k0 < EMB; k0 += KC) {
    const int buf = (k0 / KC) & 1;
    stage(buf ^ 1, k0 + KC);           // k0+KC <= EMB < KCAT: always valid
    compute(buf, Ecur, k0);
    stage_fence();
  }
  // Phase 2: K in [EMB, KCAT) -> A from h_{t-1}.
  for (int k0 = EMB; k0 < KCAT; k0 += KC) {
    const int buf = (k0 / KC) & 1;
    if (k0 + KC < KCAT) stage(buf ^ 1, k0 + KC);
    compute(buf, Hin, k0 - EMB);
    stage_fence();
  }

  // Fused LSTM cell update on the accumulators.
  // C/D layout: VGPR r, lanes 0-15 -> m=r, lanes 16-31 -> m=r+8; n=lane%16.
  const int n = j0 + (lane & 15);
  const float Bi = bi[n], Bf = bf[n], Bo = bo[n], Bg = bg[n];
#pragma unroll
  for (int mt = 0; mt < 2; ++mt) {
#pragma unroll
    for (int r = 0; r < 8; ++r) {
      const int b = b0 + mt * 16 + r + ((lane >> 4) << 3);
      const float iv = sigmoidf_(acc[0][mt][r] + Bi);
      const float fv = sigmoidf_(acc[1][mt][r] + Bf);
      const float ov = sigmoidf_(acc[2][mt][r] + Bo);
      const float gv = tanhf(acc[3][mt][r] + Bg);
      const size_t idx = (size_t)b * HID + n;
      const float c = fv * Cst[idx] + iv * gv;
      Cst[idx] = c;
      Hout[idx] = (_Float16)(ov * tanhf(c));
    }
  }
}

// ---------------------------------------------------------------------------
// Output projection: out[b][n] = hsum[b][:] . W_out[n][:] + b_out[n]
// M=256, N=32000, K=1024. Block: 256 threads = 8 waves; a 32x1024 f16 A panel
// (66KB padded) is staged once in LDS; each wave computes 2(N) x 2(M) tiles
// streaming W_out fragments from global -> 1 global frag load per 2 WMMAs.
// Grid: (N_CLASS/256, BATCH/32).
// ---------------------------------------------------------------------------
__global__ void __launch_bounds__(256)
proj_kernel(const _Float16* __restrict__ Hf,    // [BATCH][HID] f16
            const _Float16* __restrict__ Wout,  // [N_CLASS][HID] f16
            const float* __restrict__ bout,     // [N_CLASS]
            float* __restrict__ out) {          // [BATCH][N_CLASS]
  __shared__ _Float16 lA[32 * PROJ_P];

  const int m0 = blockIdx.y * 32;
  // Stage A panel: 32 rows x 1024 halves, 8 halves per store.
  for (int i = threadIdx.x; i < 32 * (HID / 8); i += 256) {
    const int row = i >> 7;            // /128
    const int col = (i & 127) << 3;    // *8
    *reinterpret_cast<v8h*>(&lA[row * PROJ_P + col]) =
        *reinterpret_cast<const v8h*>(&Hf[(size_t)(m0 + row) * HID + col]);
  }
  __syncthreads();

  const int wave = threadIdx.x >> 5;
  const int lane = threadIdx.x & 31;
  const int n0 = blockIdx.x * 256 + wave * 32;   // this wave: n0, n0+16

  v8f acc[2][2];   // [n-tile][m-tile]
#pragma unroll
  for (int i = 0; i < 2; ++i)
#pragma unroll
    for (int j = 0; j < 2; ++j) acc[i][j] = (v8f){};

  for (int k0 = 0; k0 < HID; k0 += 32) {
    v16h a0 = load_frag16(&lA[k0], PROJ_P);                 // LDS, rows 0..15
    v16h a1 = load_frag16(&lA[16 * PROJ_P + k0], PROJ_P);   // LDS, rows 16..31
#pragma unroll
    for (int nt = 0; nt < 2; ++nt) {
      v16h w = load_frag16(Wout + (size_t)(n0 + nt * 16) * HID + k0, HID);
      acc[nt][0] = __builtin_amdgcn_wmma_f32_16x16x32_f16(
          false, a0, false, w, (short)0, acc[nt][0], false, false);
      acc[nt][1] = __builtin_amdgcn_wmma_f32_16x16x32_f16(
          false, a1, false, w, (short)0, acc[nt][1], false, false);
    }
  }

  const int nn = lane & 15;
#pragma unroll
  for (int nt = 0; nt < 2; ++nt) {
    const int n = n0 + nt * 16 + nn;
    const float bv = bout[n];
#pragma unroll
    for (int mt = 0; mt < 2; ++mt) {
#pragma unroll
      for (int r = 0; r < 8; ++r) {
        const int b = m0 + mt * 16 + r + ((lane >> 4) << 3);
        out[(size_t)b * N_CLASS + n] = acc[nt][mt][r] + bv;
      }
    }
  }
}

// ---------------------------------------------------------------------------
// Host orchestration
// ---------------------------------------------------------------------------
extern "C" void kernel_launch(void* const* d_in, const int* in_sizes, int n_in,
                              void* d_out, int out_size, void* d_ws, size_t ws_size,
                              hipStream_t stream) {
  (void)in_sizes; (void)n_in; (void)out_size; (void)ws_size;

  const int*   X    = (const int*)d_in[0];
  const float* Cemb = (const float*)d_in[1];
  const float* Wg1[4] = { (const float*)d_in[2], (const float*)d_in[3],
                          (const float*)d_in[4], (const float*)d_in[5] };
  const float* Wg2[4] = { (const float*)d_in[6], (const float*)d_in[7],
                          (const float*)d_in[8], (const float*)d_in[9] };
  const float* Bv[8]  = { (const float*)d_in[10], (const float*)d_in[11],
                          (const float*)d_in[12], (const float*)d_in[13],
                          (const float*)d_in[14], (const float*)d_in[15],
                          (const float*)d_in[16], (const float*)d_in[17] };
  const float* W_out = (const float*)d_in[18];
  const float* b_out = (const float*)d_in[19];
  float* out = (float*)d_out;

  // Workspace carve-up (all chunks multiples of 256 B).
  char* ws = (char*)d_ws;
  size_t off = 0;
  _Float16* Ef16  = (_Float16*)(ws + off); off += (size_t)SEQ * BATCH * EMB * 2;
  _Float16* Wcat1 = (_Float16*)(ws + off); off += (size_t)NGATE * KCAT * 2;
  _Float16* Wcat2 = (_Float16*)(ws + off); off += (size_t)NGATE * KCAT * 2;
  _Float16* WoutH = (_Float16*)(ws + off); off += (size_t)N_CLASS * HID * 2;
  _Float16* H0    = (_Float16*)(ws + off); off += (size_t)BATCH * HID * 2;
  _Float16* H1    = (_Float16*)(ws + off); off += (size_t)BATCH * HID * 2;
  float*    Cst   = (float*)(ws + off);    off += (size_t)BATCH * HID * 4;
  float*    hsum  = (float*)(ws + off);    off += (size_t)BATCH * HID * 4;
  _Float16* HsumH = (_Float16*)(ws + off); off += (size_t)BATCH * HID * 2;

  const size_t NBH = (size_t)BATCH * HID;  // 262144

  // 1) Embedding gather + f32->f16 (time-major).
  embed_kernel<<<dim3(SEQ, BATCH), 256, 0, stream>>>(X, Cemb, Ef16);

  // 2) Pack recurrent weights (n-major fp16) and W_out.
  {
    const size_t n = (size_t)NGATE * KCAT;
    pack_gates_kernel<<<(unsigned)((n + 255) / 256), 256, 0, stream>>>(
        Wg1[0], Wg1[1], Wg1[2], Wg1[3], Wcat1);
    pack_gates_kernel<<<(unsigned)((n + 255) / 256), 256, 0, stream>>>(
        Wg2[0], Wg2[1], Wg2[2], Wg2[3], Wcat2);
    const size_t nw = (size_t)N_CLASS * HID;
    cvt_f32_to_f16_kernel<<<(unsigned)((nw + 255) / 256), 256, 0, stream>>>(
        W_out, WoutH, nw);
  }

  // 3) hsum = 0.
  zero_f32_kernel<<<(unsigned)((NBH + 255) / 256), 256, 0, stream>>>(hsum, NBH);

  // 4) Two LSTM directions (dir 1 reversed sequence).
  for (int dir = 0; dir < 2; ++dir) {
    const _Float16* Wcat = (dir == 0) ? Wcat1 : Wcat2;
    const float* bi = Bv[dir * 4 + 0];
    const float* bf = Bv[dir * 4 + 1];
    const float* bo = Bv[dir * 4 + 2];
    const float* bg = Bv[dir * 4 + 3];

    zero_f16_kernel<<<(unsigned)((NBH + 255) / 256), 256, 0, stream>>>(H0, NBH);
    zero_f16_kernel<<<(unsigned)((NBH + 255) / 256), 256, 0, stream>>>(H1, NBH);
    zero_f32_kernel<<<(unsigned)((NBH + 255) / 256), 256, 0, stream>>>(Cst, NBH);

    _Float16* Hin = H0;
    _Float16* Hout = H1;
    for (int t = 0; t < SEQ; ++t) {
      const int tt = (dir == 0) ? t : (SEQ - 1 - t);
      const _Float16* Ecur = Ef16 + (size_t)tt * BATCH * EMB;
      lstm_step_kernel<<<dim3(BATCH / 128, HID / 16), 128, 0, stream>>>(
          Ecur, Hin, Hout, Cst, Wcat, bi, bf, bo, bg);
      _Float16* tmp = Hin; Hin = Hout; Hout = tmp;
    }
    accum_h_kernel<<<(unsigned)((NBH + 255) / 256), 256, 0, stream>>>(hsum, Hin, NBH);
  }

  // 5) hsum -> f16, then output projection with bias.
  cvt_f32_to_f16_kernel<<<(unsigned)((NBH + 255) / 256), 256, 0, stream>>>(
      hsum, HsumH, NBH);
  proj_kernel<<<dim3(N_CLASS / 256, BATCH / 32), 256, 0, stream>>>(
      HsumH, WoutH, b_out, out);
}
